// EquivariantBiLinear_58737972740130
// MI455X (gfx1250) — compile-verified, matrix operand order checked by reading, not run on hardware
//
#include <hip/hip_runtime.h>
#include <stdint.h>

// ---------------------------------------------------------------------------
// out[b,o] = 0.1 * sum_{j,i} S[o,j,i] x[b,j] x[b,i]
//   with S[o,j,i] = sum_{nu : bid_g[nu*r+rho]==j} W_g[mu,nu],
//   (g,mu,rho) decoded from q = W_invperm[o*256+i].
// Phase 1: build S (f16, layout S[j][o*256+i], 33.5MB in d_ws, L2-resident).
// Phase 2: WMMA GEMM  M=256(o) x N=2048(b), K = 256(j)*256(i),
//          B-matrix G[(j,i),b] = x[b,j]*x[b,i] built on the fly from LDS.
// ---------------------------------------------------------------------------

typedef __attribute__((ext_vector_type(16))) _Float16     v16h;
typedef __attribute__((ext_vector_type(8)))  float        v8f;
typedef __attribute__((ext_vector_type(2)))  _Float16     f16x2;
typedef __attribute__((ext_vector_type(4)))  unsigned int u32x4v;
typedef __attribute__((ext_vector_type(4)))  float        f32x4v;

union Frag {
  v16h   v;
  u32x4v q[2];   // two 16B halves (8 f16 each), matches A/B K-run layout
  f16x2  p[8];
};

__device__ __forceinline__ v8f wmma_f16(v16h a, v16h b, v8f c) {
  return __builtin_amdgcn_wmma_f32_16x16x32_f16(
      /*neg_a=*/false, a, /*neg_b=*/false, b,
      /*c_mod=*/(short)0, c, /*reuse_a=*/false, /*reuse_b=*/false);
}

// ---------------------------------------------------------------------------
// Kernel 0: zero the (atomically accumulated) output. 2048*256 f32 = 2MB.
// grid 512 x 256 threads, one float4 per thread.
// ---------------------------------------------------------------------------
__global__ __launch_bounds__(256) void zero_out_kernel(float* __restrict__ out) {
  size_t i = (size_t)blockIdx.x * 256 + threadIdx.x;
  f32x4v z = {0.f, 0.f, 0.f, 0.f};
  ((f32x4v*)out)[i] = z;
}

// ---------------------------------------------------------------------------
// Kernel 1: build S. One block handles 64 consecutive p = o*256+i positions;
// 4 sub-threads share each p's nu-loop via LDS float atomics (ds_add_f32).
// acc layout [j][pl] (j*64+pl): bank = pl -> conflict-free within a wave.
// Store phase is fully coalesced: S[j*65536 + p].
// grid 1024 x 256 threads.
// ---------------------------------------------------------------------------
__global__ __launch_bounds__(256) void build_s_kernel(
    const float* __restrict__ W0, const float* __restrict__ W1,
    const float* __restrict__ W2, const float* __restrict__ W3,
    const int* __restrict__ bid0, const int* __restrict__ bid1,
    const int* __restrict__ bid2, const int* __restrict__ bid3,
    const int* __restrict__ invperm, _Float16* __restrict__ S) {
  __shared__ float acc[256 * 64];  // 64 KB

  const int t   = threadIdx.x;
  const int pl  = t & 63;   // local p
  const int sub = t >> 6;   // 0..3: quarter of the nu range

  // zero: each thread owns 64 distinct (j,pl) slots
  for (int jj = 0; jj < 64; ++jj) acc[((jj << 2) + sub) * 64 + pl] = 0.f;
  __syncthreads();

  const int p = blockIdx.x * 64 + pl;
  const int q = invperm[p];
  const int g = q >> 14;             // groups are 16384 wide each
  const int local = q & 16383;
  const int n    = 512 >> g;         // 512,256,128,64
  const int rlog = g << 1;           // 0,2,4,6
  const int mu   = local >> rlog;
  const int rho  = local & ((1 << rlog) - 1);

  const float* W; const int* bid;
  switch (g) {
    case 0:  W = W0; bid = bid0; break;
    case 1:  W = W1; bid = bid1; break;
    case 2:  W = W2; bid = bid2; break;
    default: W = W3; bid = bid3; break;
  }
  const float* wrow = W + (size_t)mu * n;

  for (int nu = sub; nu < n; nu += 4) {
    int j = bid[(nu << rlog) + rho];
    atomicAdd(&acc[(j << 6) + pl], wrow[nu]);   // ds_add_f32, bank=pl
  }
  __syncthreads();

  // coalesced store, 0.1 scale folded into S
  const size_t pbase = (size_t)blockIdx.x * 64;
  for (int jj = 0; jj < 64; ++jj) {
    int j = (jj << 2) + sub;
    float v = 0.1f * acc[(j << 6) + pl];
    S[(size_t)j * 65536 + pbase + pl] = (_Float16)v;
  }
}

// ---------------------------------------------------------------------------
// Kernel 2: WMMA GEMM.
// Block: 256 threads = 8 waves, wave grid 4(M) x 2(N), wave tile 32x32.
// Block tile: Mtile=128 (o), Ntile=64 (b), j-chunk=16 (split-K over j).
// grid = (32 Nblk, 2 Mblk, 16 jchunk) = 1024 blocks.
// LDS: x tile f16 [64 batches][256 i], padded stride 264 halves. One barrier.
// A fragments stream from global (L0/L2 resident S); B fragments are
// x[b,i] runs scaled by x[b,j] with v_pk_mul_f16.
// ---------------------------------------------------------------------------
#define XS_STRIDE 264

__global__ __launch_bounds__(256) void gemm_kernel(
    const _Float16* __restrict__ S, const float* __restrict__ x,
    float* __restrict__ out) {
  __shared__ _Float16 xs[64 * XS_STRIDE];  // ~33 KB

  const int t    = threadIdx.x;
  const int lane = t & 31;
  const int w    = t >> 5;
  const int mw   = w & 3;    // 0..3 -> M sub-tile
  const int nw   = w >> 2;   // 0..1 -> N sub-tile
  const int bN   = blockIdx.x;  // batch block (64 wide)
  const int bM   = blockIdx.y;  // o block (128 wide)
  const int j0   = blockIdx.z * 16;

  // ---- stage x tile: xs[db][i] = f16(x[bN*64+db][i]) ----
  {
    const int db  = t >> 2;
    const int ib  = (t & 3) * 64;
    const float* xp = x + (size_t)(bN * 64 + db) * 256 + ib;
    _Float16* dst   = xs + db * XS_STRIDE + ib;
#pragma unroll
    for (int k = 0; k < 64; k += 4) {
      f32x4v v = *(const f32x4v*)(xp + k);
      dst[k + 0] = (_Float16)v[0];
      dst[k + 1] = (_Float16)v[1];
      dst[k + 2] = (_Float16)v[2];
      dst[k + 3] = (_Float16)v[3];
    }
  }
  __syncthreads();

  const int hsel = lane >> 4;   // K-half select (A & B), M-half select (C/D)
  const int l16  = lane & 15;

  v8f c[2][2];
#pragma unroll
  for (int mf = 0; mf < 2; ++mf)
#pragma unroll
    for (int nf = 0; nf < 2; ++nf)
#pragma unroll
      for (int r = 0; r < 8; ++r) c[mf][nf][r] = 0.f;

  const int obase = bM * 128 + mw * 32;         // global o base for this wave
  const _Float16* xrow0 = xs + (nw * 32 +      l16) * XS_STRIDE;  // nf=0 lane col
  const _Float16* xrow1 = xs + (nw * 32 + 16 + l16) * XS_STRIDE;  // nf=1 lane col

  for (int jj = 0; jj < 16; ++jj) {
    const int j = j0 + jj;
    const _Float16 xj0s = xrow0[j];
    const _Float16 xj1s = xrow1[j];
    const f16x2 xj0 = {xj0s, xj0s};
    const f16x2 xj1 = {xj1s, xj1s};
    const _Float16* Sj = S + (size_t)j * 65536;

#pragma unroll
    for (int ks = 0; ks < 8; ++ks) {
      const int i0 = ks * 32 + hsel * 8;  // start of this lane's first K-run

      // B fragments: contiguous 8-f16 runs of x[b,i], scaled by x[b,j]
      Frag b0, b1;
      b0.q[0] = *(const u32x4v*)(xrow0 + i0);
      b0.q[1] = *(const u32x4v*)(xrow0 + i0 + 16);
      b1.q[0] = *(const u32x4v*)(xrow1 + i0);
      b1.q[1] = *(const u32x4v*)(xrow1 + i0 + 16);
#pragma unroll
      for (int k = 0; k < 8; ++k) {
        b0.p[k] *= xj0;   // v_pk_mul_f16
        b1.p[k] *= xj1;
      }

#pragma unroll
      for (int mf = 0; mf < 2; ++mf) {
        const _Float16* arow =
            Sj + (size_t)(obase + mf * 16 + l16) * 256 + i0;
        Frag a;
        a.q[0] = *(const u32x4v*)(arow);         // global b128, 16B aligned
        a.q[1] = *(const u32x4v*)(arow + 16);
        c[mf][0] = wmma_f16(a.v, b0.v, c[mf][0]);
        c[mf][1] = wmma_f16(a.v, b1.v, c[mf][1]);
      }
    }
  }

  // ---- epilogue: split-K (j-chunks) accumulate with f32 global atomics ----
#pragma unroll
  for (int mf = 0; mf < 2; ++mf)
#pragma unroll
    for (int nf = 0; nf < 2; ++nf) {
      const int o = obase + mf * 16 + hsel * 8;          // + r below
      const int b = bN * 64 + nw * 32 + nf * 16 + l16;
      float* op = out + (size_t)b * 256 + o;             // 8 consecutive o's
#pragma unroll
      for (int r = 0; r < 8; ++r) unsafeAtomicAdd(op + r, c[mf][nf][r]);
    }
}

// ---------------------------------------------------------------------------
// d_in order (setup_inputs dict order):
//  0:x  1:W0 2:bid0 3:W1 4:bid1 5:W2 6:bid2 7:W3 8:bid3 9:W_invperm
// d_ws: needs 65536*256*2 = 32 MiB for S (f16).
// ---------------------------------------------------------------------------
extern "C" void kernel_launch(void* const* d_in, const int* in_sizes, int n_in,
                              void* d_out, int out_size, void* d_ws,
                              size_t ws_size, hipStream_t stream) {
  const float* x       = (const float*)d_in[0];
  const float* W0      = (const float*)d_in[1];
  const int*   bid0    = (const int*)d_in[2];
  const float* W1      = (const float*)d_in[3];
  const int*   bid1    = (const int*)d_in[4];
  const float* W2      = (const float*)d_in[5];
  const int*   bid2    = (const int*)d_in[6];
  const float* W3      = (const float*)d_in[7];
  const int*   bid3    = (const int*)d_in[8];
  const int*   invperm = (const int*)d_in[9];
  float*       out     = (float*)d_out;
  _Float16*    S       = (_Float16*)d_ws;

  zero_out_kernel<<<512, 256, 0, stream>>>(out);
  build_s_kernel<<<1024, 256, 0, stream>>>(W0, W1, W2, W3, bid0, bid1, bid2,
                                           bid3, invperm, S);
  dim3 grid(32, 2, 16);
  gemm_kernel<<<grid, 256, 0, stream>>>(S, x, out);
}
